// PBiAttention_82085414961223
// MI455X (gfx1250) — matrix operand order repaired and checked by reading, not compiled, not run on hardware
//
#include <hip/hip_runtime.h>
#include <hip/hip_bf16.h>

// ---------------- CDNA5 WMMA types ----------------
typedef __attribute__((ext_vector_type(16))) _Float16 v16h;
typedef __attribute__((ext_vector_type(8)))  _Float16 v8h;
typedef __attribute__((ext_vector_type(8)))  float    v8f;

#define B_  2
#define C_  256
#define NH_ 32
#define HD_ 8
#define D_  16
#define N_  4096          // D*H*W
#define KTOP_ 512
#define SPA_K_ 7

__device__ __forceinline__ v16h zero_v16h() {
    v16h v;
#pragma unroll
    for (int j = 0; j < 16; ++j) v[j] = (_Float16)0.f;
    return v;
}

__device__ __forceinline__ v8f wmma16(v16h a, v16h b, v8f c) {
    return __builtin_amdgcn_wmma_f32_16x16x32_f16(
        /*neg_a=*/false, a, /*neg_b=*/false, b,
        /*c_mod=*/(short)0, c, /*reuse_a=*/false, /*reuse_b=*/false);
}

// ---------------- 1) channel mean / max -> mm[B][2][N] ----------------
__global__ void k_meanmax(const float* __restrict__ x, float* __restrict__ mm) {
    int i = blockIdx.x * blockDim.x + threadIdx.x;      // b*N + n
    if (i >= B_ * N_) return;
    int b = i / N_, n = i % N_;
    const float* xb = x + (size_t)b * C_ * N_ + n;
    float s = 0.f, mx = -__builtin_inff();
#pragma unroll 4
    for (int c = 0; c < C_; ++c) {
        float v = xb[(size_t)c * N_];
        s += v;
        mx = fmaxf(mx, v);
    }
    mm[(size_t)b * 2 * N_ + n]      = s * (1.f / C_);
    mm[(size_t)b * 2 * N_ + N_ + n] = mx;
}

// ---------------- 2) 7x7x7 conv (2ch -> 1) + sigmoid -> scores[B][N] ---------
__global__ void k_spa(const float* __restrict__ mm, const float* __restrict__ w,
                      float* __restrict__ scores) {
    int i = blockIdx.x * blockDim.x + threadIdx.x;
    if (i >= B_ * N_) return;
    int b = i / N_, n = i % N_;
    int d = n >> 8, h = (n >> 4) & 15, ww = n & 15;
    float acc = 0.f;
    for (int ci = 0; ci < 2; ++ci) {
        const float* mb = mm + (size_t)b * 2 * N_ + (size_t)ci * N_;
        const float* wc = w + ci * 343;
        for (int kd = 0; kd < SPA_K_; ++kd) {
            int zd = d + kd - 3; if (zd < 0 || zd >= D_) continue;
            for (int kh = 0; kh < SPA_K_; ++kh) {
                int zh = h + kh - 3; if (zh < 0 || zh >= D_) continue;
                for (int kw = 0; kw < SPA_K_; ++kw) {
                    int zw = ww + kw - 3; if (zw < 0 || zw >= D_) continue;
                    acc += mb[(zd << 8) + (zh << 4) + zw] * wc[kd * 49 + kh * 7 + kw];
                }
            }
        }
    }
    scores[(size_t)b * N_ + n] = 1.f / (1.f + __expf(-acc));
}

// ---------------- 3) per-batch top-512 via in-LDS bitonic sort ----------------
__global__ __launch_bounds__(1024) void k_topk(const float* __restrict__ scores,
                                               int* __restrict__ idx) {
    __shared__ float key[N_];
    __shared__ int   sid[N_];
    int b = blockIdx.x, t = threadIdx.x;
    for (int i = t; i < N_; i += 1024) {
        key[i] = scores[(size_t)b * N_ + i];
        sid[i] = i;
    }
    __syncthreads();
    for (int k2 = 2; k2 <= N_; k2 <<= 1) {
        for (int j = k2 >> 1; j > 0; j >>= 1) {
            for (int i = t; i < N_; i += 1024) {
                int ixj = i ^ j;
                if (ixj > i) {
                    bool desc = ((i & k2) == 0);
                    float a = key[i], c = key[ixj];
                    bool sw = desc ? (a < c) : (a > c);
                    if (sw) {
                        key[i] = c; key[ixj] = a;
                        int tp = sid[i]; sid[i] = sid[ixj]; sid[ixj] = tp;
                    }
                }
            }
            __syncthreads();
        }
    }
    if (t < KTOP_) idx[b * KTOP_ + t] = sid[t];
}

// ---------------- 4) x [B,C,N] -> xt f16 [B,N,C] ----------------
__global__ void k_xt(const float* __restrict__ x, _Float16* __restrict__ xt) {
    size_t i = (size_t)blockIdx.x * blockDim.x + threadIdx.x;
    if (i >= (size_t)B_ * C_ * N_) return;
    size_t b = i / ((size_t)C_ * N_);
    size_t r = i % ((size_t)C_ * N_);
    size_t c = r / N_, n = r % N_;
    xt[(b * N_ + n) * C_ + c] = (_Float16)x[i];
}

// ---------------- 5) generic f32 -> f16 ----------------
__global__ void k_cvt(const float* __restrict__ s, _Float16* __restrict__ d, int n) {
    int i = blockIdx.x * blockDim.x + threadIdx.x;
    if (i < n) d[i] = (_Float16)s[i];
}

// ---------------- 6) WMMA GEMM: C[M,Nc] = A[M,K] * W[Nc,K]^T (+bias) ----------
// N-blocked x4: each wave computes a 16x64 strip, A fragment reused across
// 4 B fragments -> 4 WMMAs per 10 b128 loads per k-step.
__global__ void k_gemm(const _Float16* __restrict__ A, const _Float16* __restrict__ Wt,
                       const float* __restrict__ bias, float* __restrict__ Cf,
                       int M, int Nc, int K, int ntiles) {
    int lane = threadIdx.x & 31;
    int warp = threadIdx.x >> 5;
    int tile = blockIdx.x * (blockDim.x >> 5) + warp;
    if (tile >= ntiles) return;
    int ntx4 = Nc >> 6;                       // strips of 64 cols
    int tm = tile / ntx4, tn4 = tile % ntx4;
    int l = lane & 15, hi = lane >> 4;
    int mrow = tm * 16 + l;
    const _Float16* Arow = A + (size_t)mrow * K;
    const _Float16* Wr0 = Wt + (size_t)(tn4 * 64 +  0 + l) * K;
    const _Float16* Wr1 = Wt + (size_t)(tn4 * 64 + 16 + l) * K;
    const _Float16* Wr2 = Wt + (size_t)(tn4 * 64 + 32 + l) * K;
    const _Float16* Wr3 = Wt + (size_t)(tn4 * 64 + 48 + l) * K;
    v8f acc0 = {}, acc1 = {}, acc2 = {}, acc3 = {};
    for (int k0 = 0; k0 < K; k0 += 32) {
        int ka = k0 + (hi ? 16 : 0);
        int kb = k0 + (hi ? 16 : 0);
        v8h a0 = *(const v8h*)(Arow + ka);
        v8h a1 = *(const v8h*)(Arow + ka + 8);
        v16h a;
#pragma unroll
        for (int j = 0; j < 8; ++j) { a[j] = a0[j]; a[j + 8] = a1[j]; }
        v16h b0 = *(const v16h*)(Wr0 + kb);
        v16h b1 = *(const v16h*)(Wr1 + kb);
        v16h b2 = *(const v16h*)(Wr2 + kb);
        v16h b3 = *(const v16h*)(Wr3 + kb);
        acc0 = wmma16(a, b0, acc0);
        acc1 = wmma16(a, b1, acc1);
        acc2 = wmma16(a, b2, acc2);
        acc3 = wmma16(a, b3, acc3);
    }
    int colbase = tn4 * 64 + l;
    float bv0 = bias ? bias[colbase]      : 0.f;
    float bv1 = bias ? bias[colbase + 16] : 0.f;
    float bv2 = bias ? bias[colbase + 32] : 0.f;
    float bv3 = bias ? bias[colbase + 48] : 0.f;
#pragma unroll
    for (int r = 0; r < 8; ++r) {
        size_t row = (size_t)(tm * 16 + r + hi * 8) * Nc;
        Cf[row + colbase]      = acc0[r] + bv0;
        Cf[row + colbase + 16] = acc1[r] + bv1;
        Cf[row + colbase + 32] = acc2[r] + bv2;
        Cf[row + colbase + 48] = acc3[r] + bv3;
    }
}

// ---------------- 7) gather top-K K/V -> head-major f16 ----------------
__global__ void k_gather(const float* __restrict__ qkv, const int* __restrict__ idx,
                         _Float16* __restrict__ kg, _Float16* __restrict__ vg) {
    int i = blockIdx.x * blockDim.x + threadIdx.x;   // b*KTOP*C
    if (i >= B_ * KTOP_ * C_) return;
    int b = i / (KTOP_ * C_);
    int r = i % (KTOP_ * C_);
    int kk = r / C_, c = r % C_;
    int n = idx[b * KTOP_ + kk];
    const float* row = qkv + ((size_t)b * N_ + n) * 768;
    int h = c >> 3, d = c & 7;
    size_t o = (((size_t)b * NH_ + h) * KTOP_ + kk) * HD_ + d;
    kg[o] = (_Float16)row[256 + c];
    vg[o] = (_Float16)row[512 + c];
}

// ---------------- 8) flash attention (WMMA) ----------------
// one wave per (b, h, 16-query tile); 4 waves / block
__global__ __launch_bounds__(128) void k_attn(const float* __restrict__ qkv,
                                              const _Float16* __restrict__ kg,
                                              const _Float16* __restrict__ vg,
                                              _Float16* __restrict__ aout) {
    __shared__ float   Sb[4][16][32];
    __shared__ _Float16 Pb[4][16][32];
    __shared__ float   mrow[4][16], lrow[4][16], arow[4][16];
    int warp = threadIdx.x >> 5, lane = threadIdx.x & 31;
    int tid = blockIdx.x * 4 + warp;                 // b*8192 + h*256 + tm
    int tm = tid & 255;
    int h  = (tid >> 8) & 31;
    int b  = tid >> 13;
    int l = lane & 15, hi = lane >> 4;
    const float scale = 0.35355339059327373f;        // 8^-0.5

    // A_q: constant across k-chunks. Valid K (=d) only 0..7 -> lanes hi==0 rows.
    v16h aq = zero_v16h();
    if (hi == 0) {
        const float* qp = qkv + ((size_t)b * N_ + tm * 16 + l) * 768 + h * HD_;
#pragma unroll
        for (int j = 0; j < 8; ++j) aq[j] = (_Float16)qp[j];
    }
    if (lane < 16) { mrow[warp][lane] = -__builtin_inff(); lrow[warp][lane] = 0.f; }

    const _Float16* kgh = kg + ((size_t)b * NH_ + h) * KTOP_ * HD_;
    const _Float16* vgh = vg + ((size_t)b * NH_ + h) * KTOP_ * HD_;
    v8f o = {};

    for (int kc = 0; kc < KTOP_; kc += 32) {
        // ---- S = Q K^T for 32 keys: two WMMAs ----
        v16h bk0 = zero_v16h(), bk1 = zero_v16h();
        if (hi == 0) {
            const _Float16* p0 = kgh + (size_t)(kc + l) * HD_;
            const _Float16* p1 = kgh + (size_t)(kc + 16 + l) * HD_;
#pragma unroll
            for (int j = 0; j < 8; ++j) { bk0[j] = p0[j]; bk1[j] = p1[j]; }
        }
        v8f s0 = {}, s1 = {};
        s0 = wmma16(aq, bk0, s0);
        s1 = wmma16(aq, bk1, s1);
#pragma unroll
        for (int r = 0; r < 8; ++r) {
            int row = r + hi * 8;
            Sb[warp][row][l]      = s0[r] * scale;
            Sb[warp][row][16 + l] = s1[r] * scale;
        }
        __syncthreads();
        // ---- online softmax row stats + P (f16) ----
        if (lane < 16) {
            int row = lane;
            float m_old = mrow[warp][row], l_old = lrow[warp][row];
            float mx = m_old;
#pragma unroll
            for (int j = 0; j < 32; ++j) mx = fmaxf(mx, Sb[warp][row][j]);
            float alpha = __expf(m_old - mx);
            float sum = l_old * alpha;
#pragma unroll
            for (int j = 0; j < 32; ++j) {
                float e = __expf(Sb[warp][row][j] - mx);
                Pb[warp][row][j] = (_Float16)e;
                sum += e;
            }
            mrow[warp][row] = mx; lrow[warp][row] = sum; arow[warp][row] = alpha;
        }
        __syncthreads();
        // ---- rescale O, then O += P V ----
#pragma unroll
        for (int r = 0; r < 8; ++r) o[r] *= arow[warp][r + hi * 8];
        v16h ap;
        {
            int ko = hi * 16;
#pragma unroll
            for (int j = 0; j < 8; ++j) {
                ap[j]     = Pb[warp][l][ko + j];
                ap[j + 8] = Pb[warp][l][8 + ko + j];
            }
        }
        v16h bv = zero_v16h();
        if (l < 8) {
            int kbase = kc + (hi ? 16 : 0);
#pragma unroll
            for (int j = 0; j < 16; ++j) bv[j] = vgh[(size_t)(kbase + j) * HD_ + l];
        }
        o = wmma16(ap, bv, o);
        __syncthreads();
    }
    // ---- normalize + store f16 [B][N][C] at col h*8+d ----
    if (l < 8) {
#pragma unroll
        for (int r = 0; r < 8; ++r) {
            int row = r + hi * 8;
            float v = o[r] / lrow[warp][row];
            aout[((size_t)b * N_ + tm * 16 + row) * C_ + h * HD_ + l] = (_Float16)v;
        }
    }
}

// ---------------- 9) depthwise 3x3x3 conv on V -> vt f16 [B,N,C] -------------
__global__ void k_dwconv(const float* __restrict__ qkv, const float* __restrict__ dw_w,
                         const float* __restrict__ dw_b, _Float16* __restrict__ vt) {
    size_t i = (size_t)blockIdx.x * blockDim.x + threadIdx.x;   // b*C*N
    if (i >= (size_t)B_ * C_ * N_) return;
    int b = (int)(i / ((size_t)C_ * N_));
    int r = (int)(i % ((size_t)C_ * N_));
    int c = r / N_, n = r % N_;
    int d = n >> 8, h = (n >> 4) & 15, ww = n & 15;
    float acc = dw_b[c];
    const float* wc = dw_w + c * 27;
    for (int kd = 0; kd < 3; ++kd) {
        int zd = d + kd - 1; if (zd < 0 || zd >= D_) continue;
        for (int kh = 0; kh < 3; ++kh) {
            int zh = h + kh - 1; if (zh < 0 || zh >= D_) continue;
            for (int kw = 0; kw < 3; ++kw) {
                int zw = ww + kw - 1; if (zw < 0 || zw >= D_) continue;
                int n2 = (zd << 8) + (zh << 4) + zw;
                acc += qkv[((size_t)b * N_ + n2) * 768 + 512 + c] * wc[kd * 9 + kh * 3 + kw];
            }
        }
    }
    vt[((size_t)b * N_ + n) * C_ + c] = (_Float16)acc;
}

// ---------------- 10) final: out[b,c,n] = proj[b,n,c] + pw[b,n,c] ------------
__global__ void k_final(const float* __restrict__ proj, const float* __restrict__ pw,
                        float* __restrict__ out) {
    size_t i = (size_t)blockIdx.x * blockDim.x + threadIdx.x;   // b*C*N
    if (i >= (size_t)B_ * C_ * N_) return;
    size_t b = i / ((size_t)C_ * N_);
    size_t r = i % ((size_t)C_ * N_);
    size_t c = r / N_, n = r % N_;
    size_t src = (b * N_ + n) * C_ + c;
    out[i] = proj[src] + pw[src];
}

// ---------------- launch ----------------
extern "C" void kernel_launch(void* const* d_in, const int* in_sizes, int n_in,
                              void* d_out, int out_size, void* d_ws, size_t ws_size,
                              hipStream_t stream) {
    (void)in_sizes; (void)n_in; (void)out_size; (void)ws_size;
    const float* x      = (const float*)d_in[0];
    const float* spa_w  = (const float*)d_in[1];
    const float* qkv_w  = (const float*)d_in[2];
    const float* proj_w = (const float*)d_in[3];
    const float* proj_b = (const float*)d_in[4];
    const float* dw_w   = (const float*)d_in[5];
    const float* dw_b   = (const float*)d_in[6];
    const float* pw_w   = (const float*)d_in[7];
    const float* pw_b   = (const float*)d_in[8];
    float* out = (float*)d_out;

    char* p = (char*)d_ws;
    auto bump = [&](size_t bytes) -> void* {
        void* r = (void*)p;
        p += (bytes + 255) & ~(size_t)255;
        return r;
    };
    float*    mm      = (float*)   bump((size_t)B_ * 2 * N_ * 4);
    float*    scores  = (float*)   bump((size_t)B_ * N_ * 4);
    int*      idx     = (int*)     bump((size_t)B_ * KTOP_ * 4);
    _Float16* xt      = (_Float16*)bump((size_t)B_ * N_ * C_ * 2);
    _Float16* wqkv_h  = (_Float16*)bump((size_t)768 * 256 * 2);
    _Float16* wproj_h = (_Float16*)bump((size_t)256 * 256 * 2);
    _Float16* wpw_h   = (_Float16*)bump((size_t)256 * 256 * 2);
    float*    qkvf    = (float*)   bump((size_t)B_ * N_ * 768 * 4);
    _Float16* kg      = (_Float16*)bump((size_t)B_ * NH_ * KTOP_ * HD_ * 2);
    _Float16* vg      = (_Float16*)bump((size_t)B_ * NH_ * KTOP_ * HD_ * 2);
    _Float16* aout    = (_Float16*)bump((size_t)B_ * N_ * C_ * 2);
    float*    projout = (float*)   bump((size_t)B_ * N_ * C_ * 4);
    _Float16* vt      = (_Float16*)bump((size_t)B_ * N_ * C_ * 2);
    float*    pwout   = (float*)   bump((size_t)B_ * N_ * C_ * 4);

    const int BN  = B_ * N_;                 // 8192
    const size_t BCN = (size_t)B_ * C_ * N_; // 2097152

    k_meanmax<<<BN / 256, 256, 0, stream>>>(x, mm);
    k_spa<<<BN / 256, 256, 0, stream>>>(mm, spa_w, scores);
    k_topk<<<B_, 1024, 0, stream>>>(scores, idx);
    k_xt<<<(int)(BCN / 256), 256, 0, stream>>>(x, xt);
    k_cvt<<<(768 * 256) / 256, 256, 0, stream>>>(qkv_w, wqkv_h, 768 * 256);
    k_cvt<<<(256 * 256) / 256, 256, 0, stream>>>(proj_w, wproj_h, 256 * 256);
    k_cvt<<<(256 * 256) / 256, 256, 0, stream>>>(pw_w, wpw_h, 256 * 256);

    int tiles_qkv = (BN / 16) * (768 / 64);               // 6144 strips of 16x64
    k_gemm<<<tiles_qkv / 8, 256, 0, stream>>>(xt, wqkv_h, nullptr, qkvf,
                                              BN, 768, 256, tiles_qkv);
    k_gather<<<(B_ * KTOP_ * C_) / 256, 256, 0, stream>>>(qkvf, idx, kg, vg);
    k_attn<<<(B_ * NH_ * (N_ / 16)) / 4, 128, 0, stream>>>(qkvf, kg, vg, aout);
    k_dwconv<<<(int)(BCN / 256), 256, 0, stream>>>(qkvf, dw_w, dw_b, vt);

    int tiles_p = (BN / 16) * (256 / 64);                 // 2048 strips of 16x64
    k_gemm<<<tiles_p / 8, 256, 0, stream>>>(aout, wproj_h, proj_b, projout,
                                            BN, 256, 256, tiles_p);
    k_gemm<<<tiles_p / 8, 256, 0, stream>>>(vt, wpw_h, pw_b, pwout,
                                            BN, 256, 256, tiles_p);
    k_final<<<(int)(BCN / 256), 256, 0, stream>>>(projout, pwout, out);
}
